// CrossViewTransformer_26147760898467
// MI455X (gfx1250) — compile-verified, matrix operand order checked by reading, not run on hardware
//
#include <hip/hip_runtime.h>
#include <hip/hip_bf16.h>
#include <math.h>

// ---------------------------------------------------------------------------
// CrossViewTransformer fused attention for MI455X (gfx1250, wave32, WMMA).
//   B=4, C=256, H=W=64, N=4096, C8=32
// Pass 0a: transpose+convert topview/sideview f32 [C][N] -> bf16 [N][C]
// Pass 0b: convert Wq/Wk/Wv -> bf16
// Pass 1 : ALL projections as WMMA GEMMs. Weight operands are preloaded
//          (Q/K) or double-buffered (V) so WMMA chains never wait on a
//          just-issued load.
// Pass 2 : flash attention computed TRANSPOSED:
//            S^T = K * Q^T     (A = K tile,  B = Q^T)
//            O^T = V^T * P^T   (A = V tile,  B = P^T packed in-register)
//          per-query softmax = in-lane reduction + one v_permlanex16 swap.
// ---------------------------------------------------------------------------

typedef __bf16 bf16;
typedef __attribute__((ext_vector_type(16))) __bf16 v16bf;
typedef __attribute__((ext_vector_type(8)))  float  v8f;
typedef __attribute__((ext_vector_type(4)))  unsigned int u32x4;

#define BB 4
#define CC 256
#define NN 4096
#define C8 32

__device__ __forceinline__ bf16 f2bf(float f) { return (bf16)f; }

// Swap values between the two 16-lane halves of the wave (lane i <-> i^16).
__device__ __forceinline__ float half_swap(float x) {
#if __has_builtin(__builtin_amdgcn_permlanex16)
    int v = __float_as_int(x);
    v = __builtin_amdgcn_permlanex16(v, v, 0x76543210, 0xfedcba98, false, false);
    return __int_as_float(v);
#else
    return __shfl_xor(x, 16, 32);
#endif
}

// Assemble two 16-byte chunks into a WMMA 16-bit operand register block.
__device__ __forceinline__ v16bf load_op(const bf16* c0, const bf16* c1) {
    union { u32x4 u[2]; v16bf v; } t;
    t.u[0] = *(const u32x4*)c0;
    t.u[1] = *(const u32x4*)c1;
    return t.v;
}

// ---------------- Pass 0a: f32 [C][N] -> bf16 [N][C] transpose -------------
__global__ __launch_bounds__(256) void cvt_transpose(
    const float* __restrict__ in, bf16* __restrict__ out)
{
    __shared__ bf16 tile[32][33];
    int blk = blockIdx.x;                 // B * (C/32) * (N/32) blocks
    int nt  = blk & 127;                  // N/32 = 128
    int ct  = (blk >> 7) & 7;             // C/32 = 8
    int b   = blk >> 10;
    int tx  = threadIdx.x & 31;
    int ty  = threadIdx.x >> 5;           // 0..7

    const float* src = in + ((size_t)b * CC + ct * 32) * NN + nt * 32;
    #pragma unroll
    for (int i = 0; i < 4; ++i) {
        int c = ty + i * 8;
        tile[c][tx] = f2bf(src[(size_t)c * NN + tx]);   // coalesced in n
    }
    __syncthreads();
    bf16* dst = out + ((size_t)b * NN + nt * 32) * CC + ct * 32;
    #pragma unroll
    for (int i = 0; i < 4; ++i) {
        int n = ty + i * 8;
        dst[(size_t)n * CC + tx] = tile[tx][n];          // coalesced in c
    }
}

// ---------------- Pass 0b: f32 -> bf16 weight convert ----------------------
__global__ __launch_bounds__(256) void cvt_w(
    const float* __restrict__ w, bf16* __restrict__ o, int n)
{
    int i = blockIdx.x * blockDim.x + threadIdx.x;
    if (i < n) o[i] = f2bf(w[i]);
}

// ---------------- Pass 1: WMMA projections ---------------------------------
// One wave per 16-pixel tile. B operand = X^T columns (pixel channel vecs),
// A operand = weight-row tiles. D: M = out-channel, N = pixel.
__global__ __launch_bounds__(128) void cvt_proj_wmma(
    const bf16* __restrict__ topT, const bf16* __restrict__ sideT,
    const bf16* __restrict__ wq, const bf16* __restrict__ wk,
    const bf16* __restrict__ wv,
    const float* __restrict__ bq, const float* __restrict__ bk,
    const float* __restrict__ bv,
    bf16* __restrict__ q, bf16* __restrict__ k, bf16* __restrict__ v)
{
    const int wave  = threadIdx.x >> 5;
    const int lane  = threadIdx.x & 31;
    const int row16 = lane & 15;
    const int half  = lane >> 4;

    const int tile = blockIdx.x * 4 + wave;   // 0 .. B*256-1
    const int b    = tile >> 8;
    const int n0   = (tile & 255) * 16;

    const v8f vzero = {0.f, 0.f, 0.f, 0.f, 0.f, 0.f, 0.f, 0.f};
    v16bf Bx[8];

    // ---------------- Q (from topT): preload all weight chunks -----------
    {
        const bf16* xrow = topT + ((size_t)b * NN + n0 + row16) * CC;
        #pragma unroll
        for (int kc = 0; kc < 8; ++kc)
            Bx[kc] = load_op(xrow + kc * 32 + 8 * half,
                             xrow + kc * 32 + 16 + 8 * half);
        v16bf Aw[16];
        #pragma unroll
        for (int ot = 0; ot < 2; ++ot)
            #pragma unroll
            for (int kc = 0; kc < 8; ++kc) {
                const bf16* wrow = wq + (ot * 16 + row16) * CC + kc * 32;
                Aw[ot * 8 + kc] = load_op(wrow + 8 * half,
                                          wrow + 16 + 8 * half);
            }
        v8f D[2] = {vzero, vzero};
        #pragma unroll
        for (int kc = 0; kc < 8; ++kc)
            #pragma unroll
            for (int ot = 0; ot < 2; ++ot)
                D[ot] = __builtin_amdgcn_wmma_f32_16x16x32_bf16(
                    false, Aw[ot * 8 + kc], false, Bx[kc],
                    (short)0, D[ot], false, false);
        #pragma unroll
        for (int ot = 0; ot < 2; ++ot) {
            union { u32x4 u; bf16 h[8]; } pk;
            #pragma unroll
            for (int r = 0; r < 8; ++r)
                pk.h[r] = f2bf(D[ot][r] + bq[ot * 16 + 8 * half + r]);
            *(u32x4*)(q + ((size_t)b * NN + n0 + row16) * C8
                        + ot * 16 + 8 * half) = pk.u;
        }
    }

    // ---------------- K, V (from sideT) ----------------
    {
        const bf16* xrow = sideT + ((size_t)b * NN + n0 + row16) * CC;
        #pragma unroll
        for (int kc = 0; kc < 8; ++kc)
            Bx[kc] = load_op(xrow + kc * 32 + 8 * half,
                             xrow + kc * 32 + 16 + 8 * half);

        // K: preload all weight chunks, then chain
        {
            v16bf Aw[16];
            #pragma unroll
            for (int ot = 0; ot < 2; ++ot)
                #pragma unroll
                for (int kc = 0; kc < 8; ++kc) {
                    const bf16* wrow = wk + (ot * 16 + row16) * CC + kc * 32;
                    Aw[ot * 8 + kc] = load_op(wrow + 8 * half,
                                              wrow + 16 + 8 * half);
                }
            v8f D[2] = {vzero, vzero};
            #pragma unroll
            for (int kc = 0; kc < 8; ++kc)
                #pragma unroll
                for (int ot = 0; ot < 2; ++ot)
                    D[ot] = __builtin_amdgcn_wmma_f32_16x16x32_bf16(
                        false, Aw[ot * 8 + kc], false, Bx[kc],
                        (short)0, D[ot], false, false);
            #pragma unroll
            for (int ot = 0; ot < 2; ++ot) {
                union { u32x4 u; bf16 h[8]; } pk;
                #pragma unroll
                for (int r = 0; r < 8; ++r)
                    pk.h[r] = f2bf(D[ot][r] + bk[ot * 16 + 8 * half + r]);
                *(u32x4*)(k + ((size_t)b * NN + n0 + row16) * C8
                            + ot * 16 + 8 * half) = pk.u;
            }
        }

        // V: 16 channel tiles, double-buffered weight operands.
        {
            v16bf AwA[8], AwB[8];
            #pragma unroll
            for (int kc = 0; kc < 8; ++kc) {          // preload ct = 0
                const bf16* wrow = wv + (row16) * CC + kc * 32;
                AwA[kc] = load_op(wrow + 8 * half, wrow + 16 + 8 * half);
            }
            #pragma unroll
            for (int ct = 0; ct < 16; ++ct) {
                v16bf* cur = (ct & 1) ? AwB : AwA;
                v16bf* nxt = (ct & 1) ? AwA : AwB;
                if (ct < 15) {                        // loads for ct+1 fly
                    #pragma unroll                    // under this chain
                    for (int kc = 0; kc < 8; ++kc) {
                        const bf16* wrow =
                            wv + ((ct + 1) * 16 + row16) * CC + kc * 32;
                        nxt[kc] = load_op(wrow + 8 * half,
                                          wrow + 16 + 8 * half);
                    }
                }
                v8f D = vzero;
                #pragma unroll
                for (int kc = 0; kc < 8; ++kc)
                    D = __builtin_amdgcn_wmma_f32_16x16x32_bf16(
                        false, cur[kc], false, Bx[kc],
                        (short)0, D, false, false);
                #pragma unroll
                for (int r = 0; r < 8; ++r) {
                    int c = ct * 16 + 8 * half + r;
                    v[((size_t)b * CC + c) * NN + n0 + row16] =
                        f2bf(D[r] + bv[c]);
                }
            }
        }
    }
}

// ---------------- Pass 2: flash attention (transposed) ---------------------
__global__ __launch_bounds__(128) void cvt_attn(
    const bf16* __restrict__ qg, const bf16* __restrict__ kg,
    const bf16* __restrict__ vg, const float* __restrict__ top,
    float* __restrict__ out)
{
    const int wave  = threadIdx.x >> 5;
    const int lane  = threadIdx.x & 31;
    const int row16 = lane & 15;
    const int half  = lane >> 4;

    const int tile = blockIdx.x * 4 + wave;              // 0 .. B*256-1
    const int b    = tile >> 8;
    const int tn   = tile & 255;
    const int q0   = tn * 16;

    // Q^T as B operand (32 channels x 16 queries): column = query vector.
    const bf16* qrow = qg + ((size_t)b * NN + q0 + row16) * C8;
    const v16bf Bq = load_op(qrow + 8 * half, qrow + 16 + 8 * half);

    v8f acc[16];                                         // O^T: 16 ch-tiles
    const v8f vzero = {0.f, 0.f, 0.f, 0.f, 0.f, 0.f, 0.f, 0.f};
    #pragma unroll
    for (int i = 0; i < 16; ++i) acc[i] = vzero;

    float m = -INFINITY, l = 0.f;                        // per-query stats

    const size_t kbase_b = (size_t)b * NN;
    const size_t vbase_b = (size_t)b * CC;

    for (int kb = 0; kb < NN; kb += 32) {
        // K tiles as A operand (16 keys x 32 channels): row = key vector.
        const bf16* k0 = kg + (kbase_b + kb + row16) * C8;
        const bf16* k1 = kg + (kbase_b + kb + 16 + row16) * C8;
        v16bf A0 = load_op(k0 + 8 * half, k0 + 16 + 8 * half);
        v16bf A1 = load_op(k1 + 8 * half, k1 + 16 + 8 * half);

        // Cover next iteration's K latency (emits global_prefetch_b8).
        if (kb + 32 < NN) {
            __builtin_prefetch(k0 + 32 * C8, 0, 3);
            __builtin_prefetch(k1 + 32 * C8, 0, 3);
        }

        // S^T tiles: (M = key, N = query) -> query scores live in-lane.
        v8f S0 = __builtin_amdgcn_wmma_f32_16x16x32_bf16(
            false, A0, false, Bq, (short)0, vzero, false, false);
        v8f S1 = __builtin_amdgcn_wmma_f32_16x16x32_bf16(
            false, A1, false, Bq, (short)0, vzero, false, false);

        // --- online softmax: in-lane reduce + one cross-half swap ---------
        float mx = -INFINITY;
        #pragma unroll
        for (int r = 0; r < 8; ++r) mx = fmaxf(mx, fmaxf(S0[r], S1[r]));
        mx = fmaxf(mx, half_swap(mx));
        float nm = fmaxf(m, mx);
        float sc = __expf(m - nm);
        m = nm;

        // Pack P^T directly into B-operand layout: lane half h holds keys
        // {8h..8h+7} (from S0) and {16+8h..16+8h+7} (from S1). In-register.
        v16bf Bp = {};
        float rs = 0.f;
        #pragma unroll
        for (int r = 0; r < 8; ++r) {
            float p0 = __expf(S0[r] - nm);
            float p1 = __expf(S1[r] - nm);
            rs += p0 + p1;
            Bp[r]     = f2bf(p0);
            Bp[8 + r] = f2bf(p1);
        }
        rs += half_swap(rs);
        l = l * sc + rs;

        // Rescale accumulators only when the running max actually moved
        // (uniform scalar branch: EXEC stays all-ones for the WMMAs).
        if (__any(sc != 1.0f)) {
            #pragma unroll
            for (int ct = 0; ct < 16; ++ct)
                #pragma unroll
                for (int r = 0; r < 8; ++r) acc[ct][r] *= sc;
        }

        // O^T += V^T * P^T : A = V tile (16 channels x 32 keys), B = Bp.
        #pragma unroll
        for (int ct = 0; ct < 16; ++ct) {
            const bf16* vrow =
                vg + (vbase_b + ct * 16 + row16) * NN + kb;
            v16bf Av = load_op(vrow + 8 * half, vrow + 16 + 8 * half);
            acc[ct] = __builtin_amdgcn_wmma_f32_16x16x32_bf16(
                false, Av, false, Bp, (short)0, acc[ct], false, false);
        }
    }

    const float inv = 1.0f / l;

    // epilogue: out[b][c][n] = top[b][c][n] + O^T[c][q].
    #pragma unroll
    for (int ct = 0; ct < 16; ++ct) {
        #pragma unroll
        for (int r = 0; r < 8; ++r) {
            int c = ct * 16 + r + 8 * half;
            size_t idx = (vbase_b + c) * NN + q0 + row16;
            out[idx] = top[idx] + acc[ct][r] * inv;
        }
    }
}

// ---------------------------------------------------------------------------
extern "C" void kernel_launch(void* const* d_in, const int* in_sizes, int n_in,
                              void* d_out, int out_size, void* d_ws, size_t ws_size,
                              hipStream_t stream) {
    const float* top  = (const float*)d_in[0];
    const float* side = (const float*)d_in[1];
    const float* Wq   = (const float*)d_in[2];
    const float* bq   = (const float*)d_in[3];
    const float* Wk   = (const float*)d_in[4];
    const float* bk   = (const float*)d_in[5];
    const float* Wv   = (const float*)d_in[6];
    const float* bv   = (const float*)d_in[7];
    float* out = (float*)d_out;

    // workspace layout (bf16):
    //   q [B][N][32]   @ 0        (1 MiB)
    //   k [B][N][32]   @ 1 MiB    (1 MiB)
    //   v [B][C][N]    @ 2 MiB    (8 MiB)
    //   topT [B][N][C] @ 10 MiB   (8 MiB)
    //   sideT[B][N][C] @ 18 MiB   (8 MiB)
    //   wq/wk/wv       @ 26 MiB   (16K + 16K + 128K)
    char* ws = (char*)d_ws;
    bf16* qb    = (bf16*)(ws);
    bf16* kb    = (bf16*)(ws + (1u  << 20));
    bf16* vb    = (bf16*)(ws + (2u  << 20));
    bf16* topT  = (bf16*)(ws + (10u << 20));
    bf16* sideT = (bf16*)(ws + (18u << 20));
    bf16* wqb   = (bf16*)(ws + (26u << 20));
    bf16* wkb   = (bf16*)(ws + (26u << 20) + (16u << 10));
    bf16* wvb   = (bf16*)(ws + (26u << 20) + (32u << 10));

    // Pass 0: transpose/convert activations + convert weights
    cvt_transpose<<<BB * 8 * 128, 256, 0, stream>>>(top,  topT);
    cvt_transpose<<<BB * 8 * 128, 256, 0, stream>>>(side, sideT);
    cvt_w<<<(C8 * CC + 255) / 256, 256, 0, stream>>>(Wq, wqb, C8 * CC);
    cvt_w<<<(C8 * CC + 255) / 256, 256, 0, stream>>>(Wk, wkb, C8 * CC);
    cvt_w<<<(CC * CC + 255) / 256, 256, 0, stream>>>(Wv, wvb, CC * CC);

    // Pass 1: WMMA projections
    cvt_proj_wmma<<<(BB * 256) / 4, 128, 0, stream>>>(
        topT, sideT, wqb, wkb, wvb, bq, bk, bv, qb, kb, vb);

    // Pass 2: WMMA flash attention
    cvt_attn<<<(BB * 256) / 4, 128, 0, stream>>>(qb, kb, vb, top, out);
}
